// BiLSTM_CRF_5480378269739
// MI455X (gfx1250) — matrix operand order repaired and checked by reading, not compiled, not run on hardware
//
#include <hip/hip_runtime.h>

// ---------------------------------------------------------------------------
// BiLSTM-CRF for MI455X (gfx1250, wave32).
// Phase 1: per layer/direction, G = f16(X) @ f16(Wih)^T + (bih+bhh) via
//          v_wmma_f32_16x16x32_f16, register-blocked 32x64 per wave
//          (8 wmma per 12 b128 loads). Operands are L2-resident.
// Phase 2: sequential LSTM recurrence, one 1024-thread WG per direction
//          (fwd/bwd concurrent). G[t] rows double-buffered into LDS with
//          global_load_async_to_lds_b64 (guarded; falls back to direct loads).
// Phase 3: projection to K=12 logits (VALU).
// Phase 4: Viterbi forward + backtrace, single wave.
// ---------------------------------------------------------------------------

#define TSEQ   2048
#define EMB    512
#define HID    1024
#define HD     512          // per-direction hidden
#define G4     2048         // 4*HD gate width
#define KTAGS  12
#define START_IDX 10
#define STOP_IDX  11

typedef _Float16 v16h __attribute__((ext_vector_type(16)));
typedef _Float16 v8h  __attribute__((ext_vector_type(8)));
typedef float    v8f  __attribute__((ext_vector_type(8)));

#if defined(__has_builtin)
#if __has_builtin(__builtin_amdgcn_global_load_async_to_lds_b64) && \
    __has_builtin(__builtin_amdgcn_s_wait_asynccnt)
#define HAVE_ASYNC_LDS 1
#endif
#endif
#ifndef HAVE_ASYNC_LDS
#define HAVE_ASYNC_LDS 0
#endif

#if HAVE_ASYNC_LDS
// ROCm clang-22 signature: (v2i addrspace(1)*, v2i addrspace(3)*, Ii, Ii)
typedef int v2i_as __attribute__((vector_size(8)));
typedef __attribute__((address_space(1))) v2i_as* g_v2i_ptr;
typedef __attribute__((address_space(3))) v2i_as* l_v2i_ptr;
#endif

// ---------------------------------------------------------------- cast f32->f16
__global__ __launch_bounds__(256) void cast_f16_kernel(
    const float* __restrict__ in, _Float16* __restrict__ out, int n) {
  int i = blockIdx.x * 256 + threadIdx.x;
  if (i < n) out[i] = (_Float16)in[i];
}

// ------------------------------------------------------- gate GEMM via WMMA
// G[t, j] = sum_k A[t,k] * W[j,k] + bih[j] + bhh[j]
// A: [TSEQ, din] f16 row-major, W: [G4, din] f16 row-major, G: [TSEQ, G4] f32.
// One wave computes a 32(M) x 64(N) tile: 2 A-frags x 4 B-frags -> 8 acc.
__device__ __forceinline__ v16h load_a_frag(const _Float16* p) {
  // A 16x32 contiguous K order: lanes 0-15 K=0..15, lanes 16-31 K=16..31.
  v8h lo = *(const v8h*)p;          // 16B contiguous
  v8h hi = *(const v8h*)(p + 8);    // 16B contiguous
  return __builtin_shufflevector(lo, hi,
      0, 1, 2, 3, 4, 5, 6, 7, 8, 9, 10, 11, 12, 13, 14, 15);
}

__global__ __launch_bounds__(256) void gate_gemm_wmma(
    const _Float16* __restrict__ A, const _Float16* __restrict__ W,
    const float* __restrict__ bih, const float* __restrict__ bhh,
    float* __restrict__ Gout, int din) {
  const int lane = threadIdx.x & 31;
  const int wave = threadIdx.x >> 5;
  const int tile = blockIdx.x * 8 + wave;   // 64 M-blocks * 32 N-blocks = 2048
  const int t0 = (tile >> 5) << 5;          // M block of 32 rows
  const int j0 = (tile & 31) << 6;          // N block of 64 cols
  const int m    = lane & 15;
  const int ksel = lane >> 4;

  const _Float16* a0 = A + (size_t)(t0 + m) * din + ksel * 16;
  const _Float16* a1 = a0 + (size_t)16 * din;
  // B fragment: lane holds col j0+16b+m (= W row), K = k + ksel*16 + {0..15}
  const _Float16* b0 = W + (size_t)(j0 + m) * din + ksel * 16;
  const _Float16* b1 = b0 + (size_t)16 * din;
  const _Float16* b2 = b0 + (size_t)32 * din;
  const _Float16* b3 = b0 + (size_t)48 * din;

  v8f acc00 = {}, acc01 = {}, acc02 = {}, acc03 = {};
  v8f acc10 = {}, acc11 = {}, acc12 = {}, acc13 = {};

  for (int k = 0; k < din; k += 32) {
    v16h av0 = load_a_frag(a0 + k);
    v16h av1 = load_a_frag(a1 + k);
    v16h bv0 = *(const v16h*)(b0 + k);      // 32B contiguous
    v16h bv1 = *(const v16h*)(b1 + k);
    v16h bv2 = *(const v16h*)(b2 + k);
    v16h bv3 = *(const v16h*)(b3 + k);
    acc00 = __builtin_amdgcn_wmma_f32_16x16x32_f16(false, av0, false, bv0, (short)0, acc00, false, false);
    acc01 = __builtin_amdgcn_wmma_f32_16x16x32_f16(false, av0, false, bv1, (short)0, acc01, false, false);
    acc02 = __builtin_amdgcn_wmma_f32_16x16x32_f16(false, av0, false, bv2, (short)0, acc02, false, false);
    acc03 = __builtin_amdgcn_wmma_f32_16x16x32_f16(false, av0, false, bv3, (short)0, acc03, false, false);
    acc10 = __builtin_amdgcn_wmma_f32_16x16x32_f16(false, av1, false, bv0, (short)0, acc10, false, false);
    acc11 = __builtin_amdgcn_wmma_f32_16x16x32_f16(false, av1, false, bv1, (short)0, acc11, false, false);
    acc12 = __builtin_amdgcn_wmma_f32_16x16x32_f16(false, av1, false, bv2, (short)0, acc12, false, false);
    acc13 = __builtin_amdgcn_wmma_f32_16x16x32_f16(false, av1, false, bv3, (short)0, acc13, false, false);
  }

  // D layout: VGPR r -> row (+ r + 8*ksel), col (+ m); bias depends on col only.
  auto store_tile = [&](const v8f& acc, int mi, int bi) {
    const int col = j0 + 16 * bi + m;
    const float bias = bih[col] + bhh[col];
    float* gp = Gout + (size_t)(t0 + 16 * mi + 8 * ksel) * G4 + col;
#pragma unroll
    for (int r = 0; r < 8; ++r) gp[(size_t)r * G4] = acc[r] + bias;
  };
  store_tile(acc00, 0, 0); store_tile(acc01, 0, 1);
  store_tile(acc02, 0, 2); store_tile(acc03, 0, 3);
  store_tile(acc10, 1, 0); store_tile(acc11, 1, 1);
  store_tile(acc12, 1, 2); store_tile(acc13, 1, 3);
}

// ------------------------------------------------------------- LSTM recurrence
// blockIdx.x: 0 = fwd, 1 = bwd. 1024 threads; thread tid computes gate rows
// tid (i or f) and tid+1024 (g or o). c lives in registers of tid<512.
__global__ __launch_bounds__(1024) void lstm_seq_kernel(
    const float* __restrict__ Gf, const float* __restrict__ Gb,
    const _Float16* __restrict__ Whf, const _Float16* __restrict__ Whb,
    float* __restrict__ Xnext) {
  const int dir = blockIdx.x;
  const float* __restrict__ G      = dir ? Gb : Gf;
  const _Float16* __restrict__ Whh = dir ? Whb : Whf;
  const int colOff = dir ? HD : 0;
  const int tid = threadIdx.x;
  const int u = tid & (HD - 1);

  __shared__ _Float16 hbuf[HD];       // h_{t-1} in f16 for the GEMV
  __shared__ float gsh[2 * HD];       // f/o pre-activations handed to tid<512
#if HAVE_ASYNC_LDS
  __shared__ float gbuf[2][G4];       // double-buffered G[t] rows (async DMA)
#endif

  if (tid < HD) hbuf[tid] = (_Float16)0.f;
  float c = 0.f;
  __syncthreads();

  const _Float16* w0 = Whh + (size_t)tid * HD;             // gate row tid
  const _Float16* w1 = Whh + (size_t)(tid + 1024) * HD;    // gate row tid+1024

  int cur = 0;
#if HAVE_ASYNC_LDS
  {
    const int tfirst = dir ? (TSEQ - 1) : 0;
    __builtin_amdgcn_global_load_async_to_lds_b64(
        (g_v2i_ptr)(G + (size_t)tfirst * G4 + 2 * tid),
        (l_v2i_ptr)(&gbuf[0][2 * tid]), 0, 0);
  }
#endif

  for (int s = 0; s < TSEQ; ++s) {
    const int t = dir ? (TSEQ - 1 - s) : s;
    const int tn = dir ? (t - 1) : (t + 1);

    float acc0 = 0.f, acc1 = 0.f;
    for (int k = 0; k < HD; k += 8) {
      v8h hv = *(const v8h*)(hbuf + k);    // LDS broadcast read
      v8h wa = *(const v8h*)(w0 + k);
      v8h wb = *(const v8h*)(w1 + k);
#pragma unroll
      for (int e = 0; e < 8; ++e) {
        float h = (float)hv[e];
        acc0 += (float)wa[e] * h;
        acc1 += (float)wb[e] * h;
      }
    }

#if HAVE_ASYNC_LDS
    __builtin_amdgcn_s_wait_asynccnt(0);   // my wave's slice of gbuf[cur] landed
#endif
    __syncthreads();   // all waves: GEMV hbuf reads done + gbuf[cur] complete

    float g0, g1;
#if HAVE_ASYNC_LDS
    g0 = gbuf[cur][tid] + acc0;
    g1 = gbuf[cur][tid + 1024] + acc1;
    if (tn >= 0 && tn < TSEQ) {            // start DMA of next step's gate row
      __builtin_amdgcn_global_load_async_to_lds_b64(
          (g_v2i_ptr)(G + (size_t)tn * G4 + 2 * tid),
          (l_v2i_ptr)(&gbuf[cur ^ 1][2 * tid]), 0, 0);
    }
#else
    if (tn >= 0 && tn < TSEQ) {            // warm next row (global_prefetch_b8)
      __builtin_prefetch(G + (size_t)tn * G4 + tid, 0, 0);
    }
    g0 = G[(size_t)t * G4 + tid] + acc0;
    g1 = G[(size_t)t * G4 + tid + 1024] + acc1;
#endif

    if (tid >= HD) {                       // publish f (g0) and o (g1)
      gsh[u] = g0;
      gsh[HD + u] = g1;
    }
    __syncthreads();
    if (tid < HD) {                        // cell update: i=g0, g=g1 here
      float ig = 1.f / (1.f + __expf(-g0));
      float gg = tanhf(g1);
      float ff = 1.f / (1.f + __expf(-gsh[u]));
      float oo = 1.f / (1.f + __expf(-gsh[HD + u]));
      c = ff * c + ig * gg;
      float h = oo * tanhf(c);
      hbuf[u] = (_Float16)h;
      Xnext[(size_t)t * HID + colOff + u] = h;
    }
    __syncthreads();                       // new hbuf visible to all
    cur ^= 1;
  }
}

// --------------------------------------------------------------- projection
__global__ __launch_bounds__(256) void proj_kernel(
    const float* __restrict__ X, const float* __restrict__ Wout,
    const float* __restrict__ bout, float* __restrict__ feats) {
  int idx = blockIdx.x * 256 + threadIdx.x;    // t*K + k
  if (idx >= TSEQ * KTAGS) return;
  int t = idx / KTAGS, k = idx - t * KTAGS;
  const float* x = X + (size_t)t * HID;
  const float* w = Wout + (size_t)k * HID;
  float acc = bout[k];
  for (int i = 0; i < HID; ++i) acc += x[i] * w[i];
  feats[idx] = acc;
}

// --------------------------------------------------------------- Viterbi
__global__ __launch_bounds__(32) void viterbi_kernel(
    const float* __restrict__ feats, const float* __restrict__ trans,
    float* __restrict__ out, int* __restrict__ bptrs) {
  const int lane = threadIdx.x;
  __shared__ float fv[KTAGS];
  __shared__ float tsh[KTAGS * KTAGS];
  __shared__ float term[KTAGS];

  if (lane < KTAGS) fv[lane] = (lane == START_IDX) ? 0.f : -10000.f;
  for (int i = lane; i < KTAGS * KTAGS; i += 32) tsh[i] = trans[i];
  __syncthreads();

  for (int t = 0; t < TSEQ; ++t) {
    float best = -3.4e38f;
    int bp = 0;
    if (lane < KTAGS) {
      best = fv[0] + tsh[lane * KTAGS + 0];
      for (int p = 1; p < KTAGS; ++p) {
        float s = fv[p] + tsh[lane * KTAGS + p];
        if (s > best) { best = s; bp = p; }     // strict > keeps first argmax
      }
      bptrs[t * KTAGS + lane] = bp;
    }
    __syncthreads();
    if (lane < KTAGS) fv[lane] = best + feats[t * KTAGS + lane];
    __syncthreads();
  }

  if (lane < KTAGS) term[lane] = fv[lane] + tsh[STOP_IDX * KTAGS + lane];
  __syncthreads();
  if (lane == 0) {
    int best = 0;
    float bs = term[0];
    for (int k = 1; k < KTAGS; ++k)
      if (term[k] > bs) { bs = term[k]; best = k; }
    out[0] = bs;                               // path_score
    int tag = best;
    for (int t = TSEQ - 1; t >= 0; --t) {      // reverse scan: ys[t] = tag
      out[1 + t] = (float)tag;
      tag = bptrs[t * KTAGS + tag];
    }
  }
}

// ---------------------------------------------------------------------------
extern "C" void kernel_launch(void* const* d_in, const int* in_sizes, int n_in,
                              void* d_out, int out_size, void* d_ws, size_t ws_size,
                              hipStream_t stream) {
  (void)in_sizes; (void)n_in; (void)out_size; (void)ws_size;

  const float* sentence = (const float*)d_in[0];
  // lstm_params flattened: layer l, dir d (0=fwd,1=bwd): Wih, Whh, bih, bhh
  auto P = [&](int l, int d, int which) -> const float* {
    return (const float*)d_in[1 + l * 8 + d * 4 + which];
  };
  const float* W_out = (const float*)d_in[41];
  const float* b_out = (const float*)d_in[42];
  const float* trans = (const float*)d_in[43];

  char* ws = (char*)d_ws;
  const size_t MB = 1024ull * 1024ull;
  float*     Xa    = (float*)(ws + 0 * MB);       // [TSEQ, HID] f32 (8 MB)
  float*     Xb    = (float*)(ws + 8 * MB);       // [TSEQ, HID] f32 (8 MB)
  _Float16*  Xh    = (_Float16*)(ws + 16 * MB);   // [TSEQ, din] f16 (<=4 MB)
  _Float16*  WihH  = (_Float16*)(ws + 20 * MB);   // [G4, din] f16 (<=4 MB)
  _Float16*  WhhF  = (_Float16*)(ws + 24 * MB);   // [G4, HD] f16 (2 MB)
  _Float16*  WhhB  = (_Float16*)(ws + 26 * MB);   // [G4, HD] f16 (2 MB)
  float*     Gf    = (float*)(ws + 28 * MB);      // [TSEQ, G4] f32 (16 MB)
  float*     Gb    = (float*)(ws + 44 * MB);      // [TSEQ, G4] f32 (16 MB)
  float*     feats = (float*)(ws + 60 * MB);      // [TSEQ, KTAGS] f32
  int*       bptrs = (int*)(ws + 60 * MB + TSEQ * KTAGS * sizeof(float));

  const float* xin = sentence;
  float* xout = Xa;

  for (int l = 0; l < 5; ++l) {
    const int din = (l == 0) ? EMB : HID;

    // f16 copy of this layer's input activations
    int nx = TSEQ * din;
    cast_f16_kernel<<<(nx + 255) / 256, 256, 0, stream>>>(xin, Xh, nx);

    // forward direction input GEMM
    int nw = G4 * din;
    cast_f16_kernel<<<(nw + 255) / 256, 256, 0, stream>>>(P(l, 0, 0), WihH, nw);
    gate_gemm_wmma<<<256, 256, 0, stream>>>(Xh, WihH, P(l, 0, 2), P(l, 0, 3), Gf, din);

    // backward direction input GEMM (reuses WihH staging buffer)
    cast_f16_kernel<<<(nw + 255) / 256, 256, 0, stream>>>(P(l, 1, 0), WihH, nw);
    gate_gemm_wmma<<<256, 256, 0, stream>>>(Xh, WihH, P(l, 1, 2), P(l, 1, 3), Gb, din);

    // recurrent weights to f16
    int nr = G4 * HD;
    cast_f16_kernel<<<(nr + 255) / 256, 256, 0, stream>>>(P(l, 0, 1), WhhF, nr);
    cast_f16_kernel<<<(nr + 255) / 256, 256, 0, stream>>>(P(l, 1, 1), WhhB, nr);

    // sequential recurrence, fwd + bwd concurrently
    lstm_seq_kernel<<<2, 1024, 0, stream>>>(Gf, Gb, WhhF, WhhB, xout);

    xin = xout;
    xout = (xout == Xa) ? Xb : Xa;
  }

  proj_kernel<<<(TSEQ * KTAGS + 255) / 256, 256, 0, stream>>>(xin, W_out, b_out, feats);
  viterbi_kernel<<<1, 32, 0, stream>>>(feats, trans, (float*)d_out, bptrs);
}
